// MultiAttention_4398046511647
// MI455X (gfx1250) — compile-verified, matrix operand order checked by reading, not run on hardware
//
#include <hip/hip_runtime.h>
#include <hip/hip_bf16.h>
#include <stdint.h>

// ---------------------------------------------------------------------------
// MultiHeadAttention for MI455X (gfx1250, wave32, WMMA 16x16x32 f16)
//   B=4 S=2048 F=512 H=8 DK=DV=64 FILT=512
// Pipeline: f16 convert -> QKV proj GEMM (WMMA, reg-double-buffered) ->
// flash attention (WMMA + online softmax + async-to-LDS double buffering) ->
// output GEMM (WMMA, f32 out)
// ---------------------------------------------------------------------------

typedef __attribute__((ext_vector_type(16))) _Float16 v16h;
typedef __attribute__((ext_vector_type(8)))  float    v8f;

union Frag16 { v16h h; uint4 q[2]; };

// Load a 16x32 f16 fragment (A layout; also our B layout when the source is
// stored reduction-contiguous per row).
//   lane%16 -> row (A: M, B: N); lane/16 -> K half {0..7,16..23} vs {8..15,24..31}
// Two 16-byte transactions; all call sites guarantee 16-B alignment.
__device__ __forceinline__ v16h load_frag(const _Float16* base, int ld) {
  const int lane = threadIdx.x & 31;
  const int hi   = lane >> 4;
  const int idx  = lane & 15;
  const _Float16* row = base + (size_t)idx * ld + hi * 8;
  Frag16 f;
  f.q[0] = *(const uint4*)(row);
  f.q[1] = *(const uint4*)(row + 16);
  return f.h;
}

__device__ __forceinline__ v8f wmma16(v16h a, v16h b, v8f c) {
  return __builtin_amdgcn_wmma_f32_16x16x32_f16(false, a, false, b,
                                                (short)0, c, false, false);
}

// CDNA5 async memory->LDS copy, 16B per lane, tracked by ASYNCcnt.
__device__ __forceinline__ void async_copy_b128(uint32_t lds_off,
                                                const _Float16* gptr) {
  asm volatile("global_load_async_to_lds_b128 %0, %1, off"
               :: "v"(lds_off), "v"((uint64_t)(uintptr_t)gptr)
               : "memory");
}
__device__ __forceinline__ void wait_async0() {
  asm volatile("s_wait_asynccnt 0x0" ::: "memory");
}

// ---------------------------------------------------------------------------
// Elementwise conversion kernels
// ---------------------------------------------------------------------------
__global__ __launch_bounds__(256) void cvt_f16(const float* __restrict__ src,
                                               _Float16* __restrict__ dst, int n) {
  int i = blockIdx.x * 256 + threadIdx.x;
  if (i < n) dst[i] = (_Float16)src[i];
}

// W[h][f][dk] (f32) -> Wt[h][dk][f] (f16), H=8 F=512 DK=64
__global__ __launch_bounds__(256) void cvt_tr_w(const float* __restrict__ W,
                                                _Float16* __restrict__ Wt) {
  int i  = blockIdx.x * 256 + threadIdx.x;
  int f  = i & 511;
  int dk = (i >> 9) & 63;
  int h  = i >> 15;
  Wt[i] = (_Float16)W[((size_t)h * 512 + f) * 64 + dk];
}

// Wo[c][o] (512x512 f32) -> Wot[o][c] (f16)
__global__ __launch_bounds__(256) void cvt_tr_wo(const float* __restrict__ W,
                                                 _Float16* __restrict__ Wt) {
  int i = blockIdx.x * 256 + threadIdx.x;
  int c = i & 511, o = i >> 9;
  Wt[i] = (_Float16)W[(size_t)c * 512 + o];
}

// ---------------------------------------------------------------------------
// QKV projection: per (which,h): [8192 x 512] @ [512 x 64] + bias
// grid (8192/128, H, 3), block 256 (8 waves, each a 16x64 tile)
// which==2 (V) stores transposed [b][h][dv][s]
// Register double-buffered K-loop so WMMAs overlap the next step's loads.
// ---------------------------------------------------------------------------
__global__ __launch_bounds__(256) void proj_kernel(
    const _Float16* __restrict__ xq, const _Float16* __restrict__ xk,
    const _Float16* __restrict__ xv,
    const _Float16* __restrict__ Wqt, const _Float16* __restrict__ Wkt,
    const _Float16* __restrict__ Wvt,
    const float* __restrict__ bq, const float* __restrict__ bk,
    const float* __restrict__ bv,
    _Float16* __restrict__ Qd, _Float16* __restrict__ Kd,
    _Float16* __restrict__ Vtd) {
  const int wave = threadIdx.x >> 5;
  const int lane = threadIdx.x & 31;
  const int h     = blockIdx.y;
  const int which = blockIdx.z;

  const _Float16* x    = which == 0 ? xq : which == 1 ? xk : xv;
  const _Float16* Wt   = which == 0 ? Wqt : which == 1 ? Wkt : Wvt;
  const float*    bias = which == 0 ? bq : which == 1 ? bk : bv;

  const int m0 = blockIdx.x * 128 + wave * 16;
  v8f acc[4] = {};
  const _Float16* Arow = x + (size_t)m0 * 512;
  const _Float16* Brow = Wt + (size_t)h * 64 * 512;   // [dk][f]

  v16h a = load_frag(Arow, 512);
  v16h b0 = load_frag(Brow, 512);
  v16h b1 = load_frag(Brow + (size_t)16 * 512, 512);
  v16h b2 = load_frag(Brow + (size_t)32 * 512, 512);
  v16h b3 = load_frag(Brow + (size_t)48 * 512, 512);
#pragma unroll 4
  for (int k = 0; k < 512; k += 32) {
    v16h an = {}, c0 = {}, c1 = {}, c2 = {}, c3 = {};
    if (k + 32 < 512) {
      an = load_frag(Arow + k + 32, 512);
      c0 = load_frag(Brow + k + 32, 512);
      c1 = load_frag(Brow + (size_t)16 * 512 + k + 32, 512);
      c2 = load_frag(Brow + (size_t)32 * 512 + k + 32, 512);
      c3 = load_frag(Brow + (size_t)48 * 512 + k + 32, 512);
    }
    acc[0] = wmma16(a, b0, acc[0]);
    acc[1] = wmma16(a, b1, acc[1]);
    acc[2] = wmma16(a, b2, acc[2]);
    acc[3] = wmma16(a, b3, acc[3]);
    a = an; b0 = c0; b1 = c1; b2 = c2; b3 = c3;
  }

  const int hf = lane >> 4, col = lane & 15;
#pragma unroll
  for (int nt = 0; nt < 4; ++nt) {
    int n = nt * 16 + col;
    float bval = bias[h * 64 + n];
#pragma unroll
    for (int r = 0; r < 8; ++r) {
      int m = m0 + r + 8 * hf;
      int b = m >> 11, s = m & 2047;
      float val = acc[nt][r] + bval;
      if (which == 0)
        Qd[(((size_t)b * 8 + h) * 2048 + s) * 64 + n] = (_Float16)val;
      else if (which == 1)
        Kd[(((size_t)b * 8 + h) * 2048 + s) * 64 + n] = (_Float16)val;
      else
        Vtd[(((size_t)b * 8 + h) * 64 + n) * 2048 + s] = (_Float16)val;
    }
  }
}

// ---------------------------------------------------------------------------
// Flash attention: grid (2048/128, B*H), block 256 (8 waves x 16 queries)
// K/V^T blocks (32 keys) double-buffered in LDS via async-to-LDS copies.
// ---------------------------------------------------------------------------
__global__ __launch_bounds__(256) void attn_kernel(
    const _Float16* __restrict__ Qd, const _Float16* __restrict__ Kd,
    const _Float16* __restrict__ Vtd, _Float16* __restrict__ concat) {
  __shared__ __align__(16) _Float16 Kl[2][32 * 64];   // [t][dk]
  __shared__ __align__(16) _Float16 Vl[2][64 * 32];   // [dv][t]
  __shared__ __align__(16) _Float16 Pl[8][16 * 32];   // per-wave P, [q][t]

  const int tid  = threadIdx.x;
  const int wave = tid >> 5;
  const int lane = tid & 31;
  const int bh   = blockIdx.y;                 // b*8 + h
  const int q0   = blockIdx.x * 128 + wave * 16;

  const _Float16* Qbh = Qd + (size_t)bh * 2048 * 64;
  const _Float16* Kbh = Kd + (size_t)bh * 2048 * 64;
  const _Float16* Vbh = Vtd + (size_t)bh * 64 * 2048;

  // hardware LDS byte offsets for async copies
  const uint32_t kbase = (uint32_t)(uintptr_t)(void*)&Kl[0][0];
  const uint32_t vbase = (uint32_t)(uintptr_t)(void*)&Vl[0][0];
  const int vrow = tid >> 2, vcg = tid & 3;    // V^T: 64 rows x 4 chunks

  const v16h qf0 = load_frag(Qbh + (size_t)q0 * 64, 64);
  const v16h qf1 = load_frag(Qbh + (size_t)q0 * 64 + 32, 64);

  float mrow[8], lrow[8];
#pragma unroll
  for (int r = 0; r < 8; ++r) { mrow[r] = -1e30f; lrow[r] = 0.0f; }
  v8f acc[4] = {};

  const float scale = 0.125f;   // 1/sqrt(DK)
  const int hf = lane >> 4, col = lane & 15;

  // stage(pp, t0): each of 256 lanes moves one 16B K chunk + one 16B V chunk
  auto stage = [&](int pp, int t0) {
    async_copy_b128(kbase + (uint32_t)(pp * 4096 + tid * 16),
                    Kbh + (size_t)t0 * 64 + tid * 8);
    async_copy_b128(vbase + (uint32_t)(pp * 4096 + (vrow * 32 + vcg * 8) * 2),
                    Vbh + (size_t)vrow * 2048 + t0 + vcg * 8);
  };

  stage(0, 0);  // prologue

  for (int n = 0; n < 64; ++n) {
    const int pp = n & 1;
    wait_async0();        // my chunks of buffer pp have landed
    __syncthreads();      // everyone's chunks have landed; prev reads done
    if (n + 1 < 64) stage(pp ^ 1, (n + 1) * 32);   // overlap next block

    const _Float16* Kc = &Kl[pp][0];
    const _Float16* Vc = &Vl[pp][0];

    // ---- scores: 16 queries x 32 keys (two 16x16 C-frags, K-dim = 64) ----
    v8f s0 = {}, s1 = {};
    {
      v16h kb0 = load_frag(Kc, 64);
      v16h kb1 = load_frag(Kc + 32, 64);
      s0 = wmma16(qf0, kb0, s0);
      s0 = wmma16(qf1, kb1, s0);
      v16h kb2 = load_frag(Kc + 16 * 64, 64);
      v16h kb3 = load_frag(Kc + 16 * 64 + 32, 64);
      s1 = wmma16(qf0, kb2, s1);
      s1 = wmma16(qf1, kb3, s1);
    }

    // ---- online softmax (rows at r + 8*hf; cols across 16-lane half) -----
    float corr[8];
#pragma unroll
    for (int r = 0; r < 8; ++r) {
      float a0 = s0[r] * scale, a1 = s1[r] * scale;
      float bm = fmaxf(a0, a1);
#pragma unroll
      for (int off = 1; off < 16; off <<= 1)
        bm = fmaxf(bm, __shfl_xor(bm, off, 32));
      float mnew = fmaxf(mrow[r], bm);
      float c  = __expf(mrow[r] - mnew);
      float p0 = __expf(a0 - mnew);
      float p1 = __expf(a1 - mnew);
      float ps = p0 + p1;
#pragma unroll
      for (int off = 1; off < 16; off <<= 1) ps += __shfl_xor(ps, off, 32);
      lrow[r] = lrow[r] * c + ps;
      mrow[r] = mnew;
      corr[r] = c;
      // C-layout -> LDS (row-major 16x32) for A-fragment reload
      Pl[wave][(r + 8 * hf) * 32 + col]      = (_Float16)p0;
      Pl[wave][(r + 8 * hf) * 32 + col + 16] = (_Float16)p1;
    }
#pragma unroll
    for (int nt = 0; nt < 4; ++nt)
#pragma unroll
      for (int r = 0; r < 8; ++r) acc[nt][r] *= corr[r];

    // ---- P @ V : A = P (16x32 over keys), B = V^T rows (dv-contig in t) ---
    v16h pf = load_frag(&Pl[wave][0], 32);
#pragma unroll
    for (int nt = 0; nt < 4; ++nt) {
      v16h vb = load_frag(Vc + (size_t)(nt * 16) * 32, 32);
      acc[nt] = wmma16(pf, vb, acc[nt]);
    }
  }

  // epilogue: normalize and write f16 concat [b][s][h*64 + dv]
  const int b = bh >> 3, h = bh & 7;
#pragma unroll
  for (int nt = 0; nt < 4; ++nt) {
#pragma unroll
    for (int r = 0; r < 8; ++r) {
      int s = q0 + r + 8 * hf;
      float val = acc[nt][r] / lrow[r];
      concat[((size_t)(b * 2048 + s)) * 512 + h * 64 + nt * 16 + col] =
          (_Float16)val;
    }
  }
}

// ---------------------------------------------------------------------------
// Output projection: [8192 x 512] @ [512 x 512] + bo -> f32
// grid (8192/128, 512/64), block 256; register double-buffered K-loop.
// ---------------------------------------------------------------------------
__global__ __launch_bounds__(256) void outproj_kernel(
    const _Float16* __restrict__ concat, const _Float16* __restrict__ Wot,
    const float* __restrict__ bo, float* __restrict__ out) {
  const int wave = threadIdx.x >> 5;
  const int lane = threadIdx.x & 31;
  const int m0 = blockIdx.x * 128 + wave * 16;
  const int n0 = blockIdx.y * 64;

  v8f acc[4] = {};
  const _Float16* Arow = concat + (size_t)m0 * 512;
  const _Float16* Brow = Wot + (size_t)n0 * 512;

  v16h a = load_frag(Arow, 512);
  v16h b0 = load_frag(Brow, 512);
  v16h b1 = load_frag(Brow + (size_t)16 * 512, 512);
  v16h b2 = load_frag(Brow + (size_t)32 * 512, 512);
  v16h b3 = load_frag(Brow + (size_t)48 * 512, 512);
#pragma unroll 4
  for (int k = 0; k < 512; k += 32) {
    v16h an = {}, c0 = {}, c1 = {}, c2 = {}, c3 = {};
    if (k + 32 < 512) {
      an = load_frag(Arow + k + 32, 512);
      c0 = load_frag(Brow + k + 32, 512);
      c1 = load_frag(Brow + (size_t)16 * 512 + k + 32, 512);
      c2 = load_frag(Brow + (size_t)32 * 512 + k + 32, 512);
      c3 = load_frag(Brow + (size_t)48 * 512 + k + 32, 512);
    }
    acc[0] = wmma16(a, b0, acc[0]);
    acc[1] = wmma16(a, b1, acc[1]);
    acc[2] = wmma16(a, b2, acc[2]);
    acc[3] = wmma16(a, b3, acc[3]);
    a = an; b0 = c0; b1 = c1; b2 = c2; b3 = c3;
  }
  const int hf = lane >> 4, col = lane & 15;
#pragma unroll
  for (int nt = 0; nt < 4; ++nt) {
    int n = n0 + nt * 16 + col;
    float bval = bo[n];
#pragma unroll
    for (int r = 0; r < 8; ++r) {
      int m = m0 + r + 8 * hf;
      out[(size_t)m * 512 + n] = acc[nt][r] + bval;
    }
  }
}

// ---------------------------------------------------------------------------
extern "C" void kernel_launch(void* const* d_in, const int* in_sizes, int n_in,
                              void* d_out, int out_size, void* d_ws,
                              size_t ws_size, hipStream_t stream) {
  const float* x_q = (const float*)d_in[0];
  const float* x_k = (const float*)d_in[1];
  const float* x_v = (const float*)d_in[2];
  const float* Wq  = (const float*)d_in[3];
  const float* bq  = (const float*)d_in[4];
  const float* Wk  = (const float*)d_in[5];
  const float* bk  = (const float*)d_in[6];
  const float* Wv  = (const float*)d_in[7];
  const float* bv  = (const float*)d_in[8];
  const float* Wo  = (const float*)d_in[9];
  const float* bo  = (const float*)d_in[10];
  float* out = (float*)d_out;
  (void)in_sizes; (void)n_in; (void)out_size; (void)ws_size;

  // workspace carve-up (~61 MB total)
  char* ws = (char*)d_ws;
  size_t off = 0;
  auto take = [&](size_t elems) {
    _Float16* p = (_Float16*)(ws + off);
    off += elems * sizeof(_Float16);
    return p;
  };
  const size_t XE = (size_t)8192 * 512;
  const size_t WE = (size_t)8 * 64 * 512;
  _Float16* xqh = take(XE);
  _Float16* xkh = take(XE);
  _Float16* xvh = take(XE);
  _Float16* Wqt = take(WE);
  _Float16* Wkt = take(WE);
  _Float16* Wvt = take(WE);
  _Float16* Wot = take((size_t)512 * 512);
  _Float16* Qd  = take(XE);
  _Float16* Kd  = take(XE);
  _Float16* Vtd = take(XE);
  _Float16* Cc  = take(XE);

  cvt_f16<<<(int)(XE / 256), 256, 0, stream>>>(x_q, xqh, (int)XE);
  cvt_f16<<<(int)(XE / 256), 256, 0, stream>>>(x_k, xkh, (int)XE);
  cvt_f16<<<(int)(XE / 256), 256, 0, stream>>>(x_v, xvh, (int)XE);
  cvt_tr_w<<<(int)(WE / 256), 256, 0, stream>>>(Wq, Wqt);
  cvt_tr_w<<<(int)(WE / 256), 256, 0, stream>>>(Wk, Wkt);
  cvt_tr_w<<<(int)(WE / 256), 256, 0, stream>>>(Wv, Wvt);
  cvt_tr_wo<<<1024, 256, 0, stream>>>(Wo, Wot);

  proj_kernel<<<dim3(64, 8, 3), 256, 0, stream>>>(
      xqh, xkh, xvh, Wqt, Wkt, Wvt, bq, bk, bv, Qd, Kd, Vtd);

  attn_kernel<<<dim3(16, 32), 256, 0, stream>>>(Qd, Kd, Vtd, Cc);

  outproj_kernel<<<dim3(64, 8), 256, 0, stream>>>(Cc, Wot, bo, out);
}
